// MultimodalMamba_58643483459598
// MI455X (gfx1250) — compile-verified, matrix operand order checked by reading, not run on hardware
//
#include <hip/hip_runtime.h>
#include <hip/hip_bf16.h>
#include <math.h>

// ---------------------------------------------------------------------------
// Multimodal Mamba for MI455X (gfx1250, wave32, WMMA).
// GEMMs: pre-packed f16 operands, v_wmma_f32_16x16x32_f16, b128 fragment loads.
// Selective scan: one wave per (batch, d_inner) channel, 4 states per lane.
// ---------------------------------------------------------------------------

#define D_MODEL 256
#define D_STATE 128
#define D_CONV  4
#define D_INNER 512
#define DT_RANK 16
#define BATCH   16
#define SEQLEN  199                // 1 + 196 + 1 + 1
#define MROWS   (BATCH * SEQLEN)   // 3184
#define DBC_N   (DT_RANK + 2 * D_STATE)  // 272

typedef __attribute__((ext_vector_type(16))) _Float16 v16h;
typedef __attribute__((ext_vector_type(8)))  _Float16 v8h;
typedef __attribute__((ext_vector_type(8)))  float    v8f;

// ---------------------------------------------------------------------------
// Generic C = A @ W^T + bias (+resid).  A: (M,K) f16 row-major, W: (N,K) f16
// row-major, K % 32 == 0.  One 16x16 tile per wave, 8 M-tiles per block.
// CDNA5 16-bit fragment layout: lane's 16 halves are K = base+0..7 and
// K = base+16..23 with base = k0 + (lane/16)*8  -> two b128 loads each.
// OUT_MODE row scatter: 0: m ; 1: m*199 ; 2: (m/196)*199+1+(m%196) ;
//                       3: m*199+197 ; 4: m*199+198
// ---------------------------------------------------------------------------
template<int OUT_MODE>
__global__ void gemm_wmma_f16(const _Float16* __restrict__ A,
                              const _Float16* __restrict__ W,
                              const float* __restrict__ bias,
                              float* __restrict__ C,
                              const float* __restrict__ resid,
                              int M, int N, int K, int out_ld) {
    const int lane  = threadIdx.x & 31;
    const int wave  = threadIdx.x >> 5;
    const int tileM = blockIdx.x * 8 + wave;
    const int tileN = blockIdx.y;
    if (tileM * 16 >= M) return;

    const int row_base = tileM * 16;
    const int col_base = tileN * 16;
    const int mrow = row_base + (lane & 15);
    const int ncol = col_base + (lane & 15);

    const _Float16* ap = A + (long)mrow * K + ((lane >> 4) << 3);
    const _Float16* bp = W + (long)ncol * K + ((lane >> 4) << 3);

    v8f acc = {};
    for (int k0 = 0; k0 < K; k0 += 32) {
        const v8h a0 = *(const v8h*)(ap + k0);
        const v8h a1 = *(const v8h*)(ap + k0 + 16);
        const v8h b0 = *(const v8h*)(bp + k0);
        const v8h b1 = *(const v8h*)(bp + k0 + 16);
        const v16h a = __builtin_shufflevector(a0, a1,
            0,1,2,3,4,5,6,7,8,9,10,11,12,13,14,15);
        const v16h b = __builtin_shufflevector(b0, b1,
            0,1,2,3,4,5,6,7,8,9,10,11,12,13,14,15);
        acc = __builtin_amdgcn_wmma_f32_16x16x32_f16(
            /*neg_a=*/false, a, /*neg_b=*/false, b,
            /*c_mod=*/(short)0, acc, /*reuse_a=*/false, /*reuse_b=*/false);
    }

    const float bv = bias ? bias[ncol] : 0.0f;
#pragma unroll
    for (int r = 0; r < 8; ++r) {
        const int m = row_base + r + ((lane >> 4) << 3);
        int orow;
        if      (OUT_MODE == 0) orow = m;
        else if (OUT_MODE == 1) orow = m * SEQLEN;
        else if (OUT_MODE == 2) orow = (m / 196) * SEQLEN + 1 + (m % 196);
        else if (OUT_MODE == 3) orow = m * SEQLEN + 197;
        else                    orow = m * SEQLEN + 198;
        const long oi = (long)orow * out_ld + ncol;
        float v = acc[r] + bv;
        if (resid) v += resid[oi];
        C[oi] = v;
    }
}

// float -> f16 row-major pack
__global__ void pack_f16(const float* __restrict__ in, _Float16* __restrict__ out, int n) {
    const int i = blockIdx.x * blockDim.x + threadIdx.x;
    if (i < n) out[i] = (_Float16)in[i];
}

// img (B,1568,196) -> imgT_h (3136,1568) f16:  out[m,k] = img[b,k,p], m=b*196+p
__global__ void pack_img_f16(const float* __restrict__ img, _Float16* __restrict__ out) {
    const int idx = blockIdx.x * blockDim.x + threadIdx.x;
    if (idx >= 3136 * 1568) return;
    const int m = idx / 1568, k = idx - m * 1568;
    const int b = m / 196,   p = m - b * 196;
    out[idx] = (_Float16)img[(long)b * (1568 * 196) + (long)k * 196 + p];
}

// seq += positional encoding; also emit f16 copy for the in_proj GEMM.
__global__ void pos_add_kernel(float* __restrict__ seq, _Float16* __restrict__ seq_h) {
    const int idx = blockIdx.x * blockDim.x + threadIdx.x;
    if (idx >= MROWS * D_MODEL) return;
    const int i = idx & (D_MODEL - 1);
    const int m = idx >> 8;
    const int t = m % SEQLEN;
    const float j2  = (float)(i & ~1);
    const float ang = (float)t * __expf(-j2 * (logf(10000.0f) / (float)D_MODEL));
    const float v = seq[idx] + ((i & 1) ? __cosf(ang) : __sinf(ang));
    seq[idx]   = v;
    seq_h[idx] = (_Float16)v;
}

// depthwise causal conv (width 4) over x half of xz, then SiLU; f32 + f16 out.
__global__ void conv_silu_kernel(const float* __restrict__ xz,
                                 const float* __restrict__ conv_w,
                                 const float* __restrict__ conv_b,
                                 float* __restrict__ xconv,
                                 _Float16* __restrict__ xconv_h) {
    const int idx = blockIdx.x * blockDim.x + threadIdx.x;
    if (idx >= MROWS * D_INNER) return;
    const int d   = idx & (D_INNER - 1);
    const int row = idx >> 9;
    const int b = row / SEQLEN, t = row % SEQLEN;
    float acc = conv_b[d];
#pragma unroll
    for (int k = 0; k < D_CONV; ++k) {
        const int tt = t - (D_CONV - 1) + k;
        if (tt >= 0)
            acc += xz[(long)(b * SEQLEN + tt) * (2 * D_INNER) + d] * conv_w[d * D_CONV + k];
    }
    const float s = acc / (1.0f + __expf(-acc));   // SiLU
    xconv[idx]   = s;
    xconv_h[idx] = (_Float16)s;
}

// delta = softplus(dt @ dt_proj_w^T + dt_proj_b)  (K=16, plain VALU)
__global__ void delta_kernel(const float* __restrict__ dbc,
                             const float* __restrict__ dtw,
                             const float* __restrict__ dtb,
                             float* __restrict__ delta) {
    const int idx = blockIdx.x * blockDim.x + threadIdx.x;
    if (idx >= MROWS * D_INNER) return;
    const int d = idx & (D_INNER - 1);
    const int m = idx >> 9;
    const float* dt = dbc + (long)m * DBC_N;
    float acc = dtb[d];
#pragma unroll
    for (int r = 0; r < DT_RANK; ++r) acc += dt[r] * dtw[d * DT_RANK + r];
    delta[idx] = (acc > 20.0f) ? acc : log1pf(__expf(acc));
}

// Selective scan: wave per (b, d); lane owns 4 states s = lane*4..lane*4+3.
// Output gated y directly as f16 (feeds the out_proj WMMA GEMM).
__global__ void scan_kernel(const float* __restrict__ delta,
                            const float* __restrict__ xconv,
                            const float* __restrict__ dbc,
                            const float* __restrict__ xz,
                            const float* __restrict__ A_log,
                            const float* __restrict__ Dp,
                            _Float16* __restrict__ yt_h) {
    const int gw   = (blockIdx.x * blockDim.x + threadIdx.x) >> 5;
    const int lane = threadIdx.x & 31;
    if (gw >= BATCH * D_INNER) return;
    const int b = gw >> 9;
    const int d = gw & (D_INNER - 1);
    const int s0 = lane * 4;

    float A0[4], h[4];
#pragma unroll
    for (int j = 0; j < 4; ++j) {
        A0[j] = -__expf(A_log[(long)d * D_STATE + s0 + j]);
        h[j]  = 0.0f;
    }
    const float dpv = Dp[d];

    for (int t = 0; t < SEQLEN; ++t) {
        const long row = (long)b * SEQLEN + t;
        const float dlt = delta[row * D_INNER + d];
        const float xv  = xconv[row * D_INNER + d];
        const float dx  = dlt * xv;
        const float* brow = dbc + row * DBC_N;
        float acc = 0.0f;
#pragma unroll
        for (int j = 0; j < 4; ++j) {
            const float Bv = brow[DT_RANK + s0 + j];
            const float Cv = brow[DT_RANK + D_STATE + s0 + j];
            h[j] = __expf(dlt * A0[j]) * h[j] + dx * Bv;
            acc += h[j] * Cv;
        }
#pragma unroll
        for (int o = 16; o > 0; o >>= 1) acc += __shfl_xor(acc, o, 32);
        if (lane == 0) {
            const float y  = acc + xv * dpv;
            const float zv = xz[row * (2 * D_INNER) + D_INNER + d];
            const float sz = zv / (1.0f + __expf(-zv));
            yt_h[row * D_INNER + d] = (_Float16)(y * sz);
        }
    }
}

// ---------------------------------------------------------------------------

extern "C" void kernel_launch(void* const* d_in, const int* in_sizes, int n_in,
                              void* d_out, int out_size, void* d_ws, size_t ws_size,
                              hipStream_t stream) {
    (void)in_sizes; (void)n_in; (void)out_size; (void)ws_size;

    const float* text  = (const float*)d_in[0];
    const float* img   = (const float*)d_in[1];
    const float* firsh = (const float*)d_in[2];
    const float* lash  = (const float*)d_in[3];
    const float* pt_w  = (const float*)d_in[4];
    const float* pt_b  = (const float*)d_in[5];
    const float* pi_w  = (const float*)d_in[6];
    const float* pi_b  = (const float*)d_in[7];
    const float* pf_w  = (const float*)d_in[8];
    const float* pf_b  = (const float*)d_in[9];
    const float* pl_w  = (const float*)d_in[10];
    const float* pl_b  = (const float*)d_in[11];
    const float* in_proj_w  = (const float*)d_in[12];
    const float* conv_w     = (const float*)d_in[13];
    const float* conv_b     = (const float*)d_in[14];
    const float* x_proj_w   = (const float*)d_in[15];
    const float* dt_proj_w  = (const float*)d_in[16];
    const float* dt_proj_b  = (const float*)d_in[17];
    const float* A_log      = (const float*)d_in[18];
    const float* Dp         = (const float*)d_in[19];
    const float* out_proj_w = (const float*)d_in[20];

    // ---- workspace: f32 region then f16 region -----------------------------
    float* ws    = (float*)d_ws;
    float* seq   = ws;                               // 3184*256   = 815104
    float* xz    = seq   + (long)MROWS * D_MODEL;    // 3184*1024  = 3260416
    float* xconv = xz    + (long)MROWS * 2*D_INNER;  // 3184*512   = 1630208
    float* dbc   = xconv + (long)MROWS * D_INNER;    // 3184*272   = 866048
    float* delta = dbc   + (long)MROWS * DBC_N;      // 3184*512   = 1630208
    _Float16* hbase = (_Float16*)(delta + (long)MROWS * D_INNER);

    _Float16* text_h  = hbase;                        // 16*768
    _Float16* img_h   = text_h  + 16L*768;            // 3136*1568
    _Float16* f_h     = img_h   + 3136L*1568;         // 16*3584
    _Float16* l_h     = f_h     + 16L*3584;           // 16*3584
    _Float16* seq_h   = l_h     + 16L*3584;           // 3184*256
    _Float16* xconv_h = seq_h   + (long)MROWS*D_MODEL;// 3184*512
    _Float16* yt_h    = xconv_h + (long)MROWS*D_INNER;// 3184*512
    _Float16* w_pt    = yt_h    + (long)MROWS*D_INNER;// 256*768
    _Float16* w_pi    = w_pt    + 256L*768;           // 256*1568
    _Float16* w_pf    = w_pi    + 256L*1568;          // 256*3584
    _Float16* w_pl    = w_pf    + 256L*3584;          // 256*3584
    _Float16* w_in    = w_pl    + 256L*3584;          // 1024*256
    _Float16* w_xp    = w_in    + 1024L*256;          // 272*512
    _Float16* w_op    = w_xp    + 272L*512;           // 256*512
    float* out = (float*)d_out;

    const dim3 blk(256);
    auto nblk = [](long n) { return dim3((unsigned)((n + 255) / 256)); };
    auto gemm_grid = [](int M, int N) { return dim3((M + 127) / 128, N / 16); };

    // ---- 0) pack operands to f16 ------------------------------------------
    pack_f16<<<nblk(16L*768),    blk, 0, stream>>>(text,       text_h, 16*768);
    pack_img_f16<<<nblk(3136L*1568), blk, 0, stream>>>(img,    img_h);
    pack_f16<<<nblk(16L*3584),   blk, 0, stream>>>(firsh,      f_h,  16*3584);
    pack_f16<<<nblk(16L*3584),   blk, 0, stream>>>(lash,       l_h,  16*3584);
    pack_f16<<<nblk(256L*768),   blk, 0, stream>>>(pt_w,       w_pt, 256*768);
    pack_f16<<<nblk(256L*1568),  blk, 0, stream>>>(pi_w,       w_pi, 256*1568);
    pack_f16<<<nblk(256L*3584),  blk, 0, stream>>>(pf_w,       w_pf, 256*3584);
    pack_f16<<<nblk(256L*3584),  blk, 0, stream>>>(pl_w,       w_pl, 256*3584);
    pack_f16<<<nblk(1024L*256),  blk, 0, stream>>>(in_proj_w,  w_in, 1024*256);
    pack_f16<<<nblk(272L*512),   blk, 0, stream>>>(x_proj_w,   w_xp, 272*512);
    pack_f16<<<nblk(256L*512),   blk, 0, stream>>>(out_proj_w, w_op, 256*512);

    // ---- 1) modality projections -> seq (row-scattered, ld = D_MODEL) -----
    gemm_wmma_f16<1><<<gemm_grid(16,   D_MODEL), blk, 0, stream>>>(
        text_h, w_pt, pt_b, seq, nullptr, 16,   D_MODEL, 768,  D_MODEL);
    gemm_wmma_f16<2><<<gemm_grid(3136, D_MODEL), blk, 0, stream>>>(
        img_h,  w_pi, pi_b, seq, nullptr, 3136, D_MODEL, 1568, D_MODEL);
    gemm_wmma_f16<3><<<gemm_grid(16,   D_MODEL), blk, 0, stream>>>(
        f_h,    w_pf, pf_b, seq, nullptr, 16,   D_MODEL, 3584, D_MODEL);
    gemm_wmma_f16<4><<<gemm_grid(16,   D_MODEL), blk, 0, stream>>>(
        l_h,    w_pl, pl_b, seq, nullptr, 16,   D_MODEL, 3584, D_MODEL);

    // ---- 2) + positional encoding (also packs seq to f16) ------------------
    pos_add_kernel<<<nblk((long)MROWS * D_MODEL), blk, 0, stream>>>(seq, seq_h);

    // ---- 3) in_proj: xz = seq @ in_proj_w^T  (3184 x 1024) -----------------
    gemm_wmma_f16<0><<<gemm_grid(MROWS, 2*D_INNER), blk, 0, stream>>>(
        seq_h, w_in, nullptr, xz, nullptr, MROWS, 2*D_INNER, D_MODEL, 2*D_INNER);

    // ---- 4) depthwise causal conv + SiLU -----------------------------------
    conv_silu_kernel<<<nblk((long)MROWS * D_INNER), blk, 0, stream>>>(
        xz, conv_w, conv_b, xconv, xconv_h);

    // ---- 5) x_proj: dbc = xconv @ x_proj_w^T  (3184 x 272) -----------------
    gemm_wmma_f16<0><<<gemm_grid(MROWS, DBC_N), blk, 0, stream>>>(
        xconv_h, w_xp, nullptr, dbc, nullptr, MROWS, DBC_N, D_INNER, DBC_N);

    // ---- 6) delta = softplus(dt @ dt_proj_w^T + b) -------------------------
    delta_kernel<<<nblk((long)MROWS * D_INNER), blk, 0, stream>>>(
        dbc, dt_proj_w, dt_proj_b, delta);

    // ---- 7) selective scan + gate -> yt (f16) ------------------------------
    scan_kernel<<<nblk((long)BATCH * D_INNER * 32), blk, 0, stream>>>(
        delta, xconv, dbc, xz, A_log, Dp, yt_h);

    // ---- 8) out_proj + residual -> d_out -----------------------------------
    gemm_wmma_f16<0><<<gemm_grid(MROWS, D_MODEL), blk, 0, stream>>>(
        yt_h, w_op, nullptr, out, seq, MROWS, D_MODEL, D_INNER, D_MODEL);
}
